// Attention_58016418235000
// MI455X (gfx1250) — matrix operand order repaired and checked by reading, not compiled, not run on hardware
//
#include <hip/hip_runtime.h>
#include <hip/hip_bf16.h>

// ---------------------------------------------------------------------------
// Seq2seq LSTM + attention for MI455X (gfx1250), wave32 + WMMA bf16 split-fp32.
//
// Pipeline:
//  1) k_pack    : pack W (row-major [N,K] f32) -> bf16 hi/lo tiles in the exact
//                 per-lane layout of V_WMMA_F32_16X16X32_BF16's B operand.
//  2) k_gather  : one-hot @ W_ih.T == column gather of W_ih  (+ folded biases).
//  3) k_lstm    : sequential recurrence, 1 workgroup of 16 waves (512 thr,
//                 4 waves/SIMD -> 256-VGPR budget, no spills). Weight-tile
//                 loads use compile-time-constant instruction offsets from one
//                 per-wave base pointer; 3 split-product WMMA terms issued as
//                 4-way round-robin sweeps so no same-acc WMMAs are adjacent.
//  4) k_gemm    : generic split-bf16 WMMA GEMM (proj_enc, hid(tanh), logits).
//  5) k_attn_*  : per-batch attention with enc state cached in 256KB LDS.
//  6) k_softmax : row softmax in-place on d_out.
// ---------------------------------------------------------------------------

typedef __attribute__((ext_vector_type(16))) __bf16 v16bf;
typedef __attribute__((ext_vector_type(8)))  float  v8f;
typedef unsigned short u16;

#define Bsz 32
#define Ssz 128
#define Tsz 128
#define Hsz 512
#define Gsz 2048   // 4*H
#define Vsz 8000

// ---- bf16 helpers (bit-level; avoid relying on __bf16 arithmetic) ----------
__device__ __forceinline__ u16 f32_to_bf16_rne(float f) {
  unsigned u = __builtin_bit_cast(unsigned, f);
  unsigned r = u + 0x7fffu + ((u >> 16) & 1u);   // round-to-nearest-even
  return (u16)(r >> 16);
}
// hi = truncated bf16, lo = bf16(residual); hi+lo ~ f to ~2^-16 rel. error
__device__ __forceinline__ void split_bf16(float f, u16& hi, u16& lo) {
  unsigned u = __builtin_bit_cast(unsigned, f);
  hi = (u16)(u >> 16);
  float fh = __builtin_bit_cast(float, (unsigned)hi << 16);
  lo = f32_to_bf16_rne(f - fh);
}
__device__ __forceinline__ __bf16 bfbits(u16 v) {
  return __builtin_bit_cast(__bf16, v);
}
__device__ __forceinline__ float sigm(float x) { return 1.0f / (1.0f + __expf(-x)); }

#define WMMA_BF16(A, Bm, C) \
  __builtin_amdgcn_wmma_f32_16x16x32_bf16(false, (A), false, (Bm), (short)0, (C), false, false)

// ---------------------------------------------------------------------------
// Pack W [N,K] f32 row-major into WMMA-B bf16 hi/lo tile layout.
// B operand (32x16, K x N), per lane l: n = l&15, g = l>>4, K(e) = g*8+e (e<8) else 16+g*8+(e-8).
// Tile index = ntile*KT + kt ; fragment = 32 lanes x 16 bf16 (32B/lane).
// ---------------------------------------------------------------------------
__global__ void k_pack(const float* __restrict__ W, u16* __restrict__ hi,
                       u16* __restrict__ lo, int N, int K) {
  int gt = blockIdx.x * blockDim.x + threadIdx.x;
  int KT = K >> 5;
  int total = (N >> 4) * KT * 32;
  if (gt >= total) return;
  int lane = gt & 31;
  int tile = gt >> 5;
  int kt = tile % KT, ntile = tile / KT;
  int n = ntile * 16 + (lane & 15);
  int g = lane >> 4;
  size_t base = (size_t)n * K + kt * 32 + g * 8;
  size_t ob = (size_t)tile * 512 + lane * 16;
#pragma unroll
  for (int e = 0; e < 16; ++e) {
    u16 h, l;
    split_bf16(W[base + e + (e >> 3) * 8], h, l);
    hi[ob + e] = h;
    lo[ob + e] = l;
  }
}

// ---------------------------------------------------------------------------
// xp[sb, g] = W_ih[g, tok(s,b)] + b_ih[g] + b_hh[g]   (one-hot GEMM == gather)
// tokens laid out [B, L] (row-major), xp laid out [L*B, G].
// ---------------------------------------------------------------------------
__global__ void k_gather(const int* __restrict__ tok, const float* __restrict__ Wih,
                         const float* __restrict__ b0, const float* __restrict__ b1,
                         float* __restrict__ xp, int L) {
  int sb = blockIdx.x;           // s*B + b
  int s = sb >> 5, b = sb & 31;
  int t = tok[b * L + s];
  for (int g = threadIdx.x; g < Gsz; g += blockDim.x)
    xp[(size_t)sb * Gsz + g] = Wih[(size_t)g * Vsz + t] + b0[g] + b1[g];
}

// ---------------------------------------------------------------------------
// Sequential LSTM. One workgroup, 512 threads = 16 waves (4 waves/SIMD,
// 256-VGPR budget per wave -> accumulators stay in registers, no spills).
// gates[32,2048] = xp_t + h[32,512] @ Whh^T.   Wave (mt, jt0): owns the four
// gate tiles (i,f,g,o) of columns j in [jt0*16, (jt0+4)*16) for batch rows
// mt*16..mt*16+15  ->  LSTM cell evaluated fully in registers.
// h is stored in LDS pre-packed in the WMMA-A bf16 layout:
//   A lane l: m = l&15, g = l>>4; K(e) = g*8+e (e<8) else 16+g*8+(e-8).
// Weight fragment element offset for tile (a,q) at step kt:
//   q*262144 + (jt0+a)*8192 + kt*512 + lane*16
// -> q,a parts become compile-time-constant instruction offsets (<8MB).
// ---------------------------------------------------------------------------
__global__ __launch_bounds__(512, 1)
void k_lstm(const float* __restrict__ xp, const u16* __restrict__ Whi,
            const u16* __restrict__ Wlo, const float* h_init, const float* c_init,
            float* __restrict__ h_seq, float* __restrict__ c_out, int L) {
  extern __shared__ char smraw[];
  u16* hhi = (u16*)smraw;          // [2 mt][16 kt][32 lane][16 e]
  u16* hlo = hhi + 16384;

  const int tid = threadIdx.x, lane = tid & 31, wid = tid >> 5;
  const int mt = wid & 1, jt0 = (wid >> 1) << 2;   // 8 wave-pairs x 4 jt = 32 jt
  const int ml = lane & 15, g = lane >> 4;

  // init packed h (zeros or h_init)
  for (int idx = tid; idx < 16384; idx += 512) {
    float v = 0.0f;
    if (h_init) {
      int e = idx & 15, ln = (idx >> 4) & 31, kt = (idx >> 9) & 15, mtv = idx >> 13;
      int m = mtv * 16 + (ln & 15);
      int gg = ln >> 4;
      int kk = (e & 7) + 8 * gg + 16 * (e >> 3);
      v = h_init[(size_t)m * Hsz + kt * 32 + kk];
    }
    u16 h, l;
    split_bf16(v, h, l);
    hhi[idx] = h;
    hlo[idx] = l;
  }
  // persistent cell state in registers (fixed D-fragment coordinates)
  float creg[4][8];
#pragma unroll
  for (int a = 0; a < 4; ++a) {
    int j = (jt0 + a) * 16 + ml;
#pragma unroll
    for (int r = 0; r < 8; ++r) {
      int m = mt * 16 + g * 8 + r;
      creg[a][r] = c_init ? c_init[(size_t)m * Hsz + j] : 0.0f;
    }
  }
  // per-wave weight base pointers (q/a offsets become instruction immediates)
  const u16* baseH = Whi + (size_t)jt0 * 8192 + lane * 16;
  const u16* baseL = Wlo + (size_t)jt0 * 8192 + lane * 16;
  __syncthreads();

  for (int step = 0; step < L; ++step) {
    // accumulators seeded with xp (biases already folded in)
    v8f acc[4][4];
#pragma unroll
    for (int a = 0; a < 4; ++a) {
      int j = (jt0 + a) * 16 + ml;
#pragma unroll
      for (int q = 0; q < 4; ++q)
#pragma unroll
        for (int r = 0; r < 8; ++r) {
          int m = mt * 16 + g * 8 + r;
          acc[a][q][r] = xp[(size_t)(step * Bsz + m) * Gsz + q * Hsz + j];
        }
    }
    // gates += h @ Whh^T  (split-bf16: 3 WMMA per tile, issued as round-robin
    // sweeps over the 4 q-accumulators so no same-acc WMMAs are adjacent)
    for (int kt = 0; kt < 16; ++kt) {
      v16bf ahi = *(const v16bf*)(hhi + ((mt * 16 + kt) * 32 + lane) * 16);
      v16bf alo = *(const v16bf*)(hlo + ((mt * 16 + kt) * 32 + lane) * 16);
      const u16* pH = baseH + kt * 512;
      const u16* pL = baseL + kt * 512;
#pragma unroll
      for (int a = 0; a < 4; ++a) {
        v16bf bh[4], bl[4];
#pragma unroll
        for (int q = 0; q < 4; ++q) {
          // constant byte offset: q*524288 + a*16384
          bh[q] = *(const v16bf*)(pH + (size_t)q * 262144 + a * 8192);
          bl[q] = *(const v16bf*)(pL + (size_t)q * 262144 + a * 8192);
        }
#pragma unroll
        for (int q = 0; q < 4; ++q) acc[a][q] = WMMA_BF16(ahi, bh[q], acc[a][q]);
#pragma unroll
        for (int q = 0; q < 4; ++q) acc[a][q] = WMMA_BF16(alo, bh[q], acc[a][q]);
#pragma unroll
        for (int q = 0; q < 4; ++q) acc[a][q] = WMMA_BF16(ahi, bl[q], acc[a][q]);
      }
    }
    __syncthreads();  // all waves done reading packed h

    // LSTM cell in registers; write h to global + re-pack into LDS
#pragma unroll
    for (int a = 0; a < 4; ++a) {
      int j = (jt0 + a) * 16 + ml;
      int kt2 = j >> 5, kk = j & 31;
      int gp = (kk >> 3) & 1;
      int e = (kk & 7) + 8 * (kk >> 4);
#pragma unroll
      for (int r = 0; r < 8; ++r) {
        int m = mt * 16 + g * 8 + r;
        float ig = sigm(acc[a][0][r]);
        float fg = sigm(acc[a][1][r]);
        float gg2 = tanhf(acc[a][2][r]);
        float og = sigm(acc[a][3][r]);
        float c = fg * creg[a][r] + ig * gg2;
        creg[a][r] = c;
        float h = og * tanhf(c);
        h_seq[(size_t)(step * Bsz + m) * Hsz + j] = h;
        int lane2 = (m & 15) + (gp << 4);
        int idx = ((mt * 16 + kt2) * 32 + lane2) * 16 + e;
        u16 hb, lb;
        split_bf16(h, hb, lb);
        hhi[idx] = hb;
        hlo[idx] = lb;
      }
    }
    __syncthreads();  // packed h complete before next step reads it
  }
  // dump final cell state
#pragma unroll
  for (int a = 0; a < 4; ++a) {
    int j = (jt0 + a) * 16 + ml;
#pragma unroll
    for (int r = 0; r < 8; ++r) {
      int m = mt * 16 + g * 8 + r;
      c_out[(size_t)m * Hsz + j] = creg[a][r];
    }
  }
}

// ---------------------------------------------------------------------------
// Generic split-bf16 WMMA GEMM:  C[M,N] = act(A[M,K] @ Wpacked^T + bias)
// Each wave owns one 16x64 output tile (1 mtile x 4 ntiles), loops K/32.
// The 3 split-product terms are issued round-robin over the 4 accumulators
// so no two consecutive WMMAs share a destination.
// remap=1: row m = t*32+b is written to C row b*128+t (logits -> d_out).
// ---------------------------------------------------------------------------
__global__ void k_gemm(const float* __restrict__ A, const u16* __restrict__ Bhi,
                       const u16* __restrict__ Blo, const float* __restrict__ bias,
                       float* __restrict__ C, int M, int N, int K, int act, int remap) {
  int lane = threadIdx.x & 31;
  int wid = (blockIdx.x * blockDim.x + threadIdx.x) >> 5;
  int Ng = N >> 6;
  if (wid >= (M >> 4) * Ng) return;
  int mt = wid / Ng, ng = wid % Ng;
  int KT = K >> 5;
  int ml = lane & 15, g = lane >> 4;

  v8f acc[4];
#pragma unroll
  for (int t = 0; t < 4; ++t)
#pragma unroll
    for (int r = 0; r < 8; ++r) acc[t][r] = 0.0f;

  const float* Arow = A + (size_t)(mt * 16 + ml) * K;
  for (int kt = 0; kt < KT; ++kt) {
    int kb = kt * 32 + g * 8;
    v16bf ahi, alo;
#pragma unroll
    for (int e = 0; e < 8; ++e) {
      u16 h, l;
      split_bf16(Arow[kb + e], h, l);
      ahi[e] = bfbits(h);
      alo[e] = bfbits(l);
    }
#pragma unroll
    for (int e = 0; e < 8; ++e) {
      u16 h, l;
      split_bf16(Arow[kb + 16 + e], h, l);
      ahi[8 + e] = bfbits(h);
      alo[8 + e] = bfbits(l);
    }
    v16bf bh[4], bl[4];
#pragma unroll
    for (int t = 0; t < 4; ++t) {
      size_t toff = ((size_t)((ng * 4 + t) * KT + kt) * 32 + lane) * 16;
      bh[t] = *(const v16bf*)(Bhi + toff);
      bl[t] = *(const v16bf*)(Blo + toff);
    }
#pragma unroll
    for (int t = 0; t < 4; ++t) acc[t] = WMMA_BF16(ahi, bh[t], acc[t]);
#pragma unroll
    for (int t = 0; t < 4; ++t) acc[t] = WMMA_BF16(alo, bh[t], acc[t]);
#pragma unroll
    for (int t = 0; t < 4; ++t) acc[t] = WMMA_BF16(ahi, bl[t], acc[t]);
  }
#pragma unroll
  for (int t = 0; t < 4; ++t) {
    int n0 = (ng * 4 + t) * 16;
    float bv = bias[n0 + ml];
#pragma unroll
    for (int r = 0; r < 8; ++r) {
      int m = mt * 16 + r + 8 * g;
      float v = acc[t][r] + bv;
      if (act) v = tanhf(v);
      int crow = remap ? ((m & 31) * Tsz + (m >> 5)) : m;
      C[(size_t)crow * N + n0 + ml] = v;
    }
  }
}

// ---------------------------------------------------------------------------
// Attention scores + softmax over s. One block per batch b; proj_enc[:,b,:]
// (256KB) staged once in LDS, reused for all 128 decoder steps.
// attw[t,b,s] = softmax_s( proj_enc[s,b,:] . dec_h[t,b,:] )
// ---------------------------------------------------------------------------
__global__ void k_attn_scores(const float* __restrict__ proj, const float* __restrict__ dec_h,
                              float* __restrict__ attw) {
  extern __shared__ float sm[];
  float* P = sm;              // 128*512
  float* dh = P + 65536;      // 512
  float* red = dh + 512;      // 256
  float* sc = red + 256;      // 128
  int b = blockIdx.x, tid = threadIdx.x;
  for (int i = tid; i < 128 * 512; i += 256) {
    int s = i >> 9, j = i & 511;
    P[i] = proj[(size_t)(s * Bsz + b) * Hsz + j];
  }
  __syncthreads();
  for (int t = 0; t < Tsz; ++t) {
    for (int i = tid; i < Hsz; i += 256) dh[i] = dec_h[(size_t)(t * Bsz + b) * Hsz + i];
    __syncthreads();
    int s = tid >> 1, half = tid & 1;
    const float* pr = P + s * Hsz + half * 256;
    const float* dd = dh + half * 256;
    float a = 0.0f;
    for (int j = 0; j < 256; ++j) a += pr[j] * dd[j];
    red[tid] = a;
    __syncthreads();
    if (!half) sc[s] = red[tid] + red[tid + 1];
    __syncthreads();
    if (tid < 128) red[tid] = sc[tid];
    __syncthreads();
    for (int o = 64; o >= 1; o >>= 1) {
      if (tid < o) red[tid] = fmaxf(red[tid], red[tid + o]);
      __syncthreads();
    }
    float mx = red[0];
    __syncthreads();
    if (tid < 128) {
      float e = __expf(sc[tid] - mx);
      sc[tid] = e;
      red[tid] = e;
    }
    __syncthreads();
    for (int o = 64; o >= 1; o >>= 1) {
      if (tid < o) red[tid] += red[tid + o];
      __syncthreads();
    }
    float inv = 1.0f / red[0];
    if (tid < 128) attw[(size_t)(t * Bsz + b) * Ssz + tid] = sc[tid] * inv;
    __syncthreads();
  }
}

// ---------------------------------------------------------------------------
// context[t,b,:] = sum_s attw[t,b,s] * enc_h[s,b,:]; also assembles the
// concat buffer hc[t*B+b] = [dec_h | context]  (input of the hid GEMM).
// enc_h[:,b,:] (256KB) staged once in LDS per block.
// ---------------------------------------------------------------------------
__global__ void k_attn_ctx(const float* __restrict__ enc_h, const float* __restrict__ dec_h,
                           const float* __restrict__ attw, float* __restrict__ hc) {
  extern __shared__ float sm[];
  float* E = sm;          // 128*512
  float* w = E + 65536;   // 128
  int b = blockIdx.x, tid = threadIdx.x;
  for (int i = tid; i < 128 * 512; i += 256) {
    int s = i >> 9, j = i & 511;
    E[i] = enc_h[(size_t)(s * Bsz + b) * Hsz + j];
  }
  __syncthreads();
  for (int t = 0; t < Tsz; ++t) {
    size_t row = (size_t)(t * Bsz + b);
    if (tid < 128) w[tid] = attw[row * Ssz + tid];
    __syncthreads();
    for (int j = tid; j < Hsz; j += 256) {
      float a = 0.0f;
#pragma unroll 4
      for (int s = 0; s < Ssz; ++s) a += w[s] * E[s * Hsz + j];
      hc[row * (2 * Hsz) + Hsz + j] = a;
      hc[row * (2 * Hsz) + j] = dec_h[row * Hsz + j];
    }
    __syncthreads();
  }
}

// ---------------------------------------------------------------------------
// In-place row softmax over VE=8000 on d_out rows (b*T + t).
// ---------------------------------------------------------------------------
__global__ void k_softmax(float* __restrict__ out) {
  __shared__ float red[256];
  float* p = out + (size_t)blockIdx.x * Vsz;
  int tid = threadIdx.x;
  float m = -3.402823466e+38f;
  for (int i = tid; i < Vsz; i += 256) m = fmaxf(m, p[i]);
  red[tid] = m;
  __syncthreads();
  for (int o = 128; o >= 1; o >>= 1) {
    if (tid < o) red[tid] = fmaxf(red[tid], red[tid + o]);
    __syncthreads();
  }
  float mx = red[0];
  __syncthreads();
  float s = 0.0f;
  for (int i = tid; i < Vsz; i += 256) {
    float e = __expf(p[i] - mx);
    p[i] = e;
    s += e;
  }
  red[tid] = s;
  __syncthreads();
  for (int o = 128; o >= 1; o >>= 1) {
    if (tid < o) red[tid] += red[tid + o];
    __syncthreads();
  }
  float inv = 1.0f / red[0];
  for (int i = tid; i < Vsz; i += 256) p[i] *= inv;
}

// ---------------------------------------------------------------------------
extern "C" void kernel_launch(void* const* d_in, const int* in_sizes, int n_in,
                              void* d_out, int out_size, void* d_ws, size_t ws_size,
                              hipStream_t stream) {
  const int* input_batch = (const int*)d_in[0];     // [B,S]
  const int* output_batch = (const int*)d_in[1];    // [B,T]
  const float* Wa = (const float*)d_in[2];          // [H,H]
  const float* ba = (const float*)d_in[3];
  const float* W1 = (const float*)d_in[4];          // [H,2H]
  const float* b1 = (const float*)d_in[5];
  const float* W2 = (const float*)d_in[6];          // [VE,H]
  const float* b2 = (const float*)d_in[7];
  const float* We_ih = (const float*)d_in[8];       // [4H,VS]
  const float* We_hh = (const float*)d_in[9];       // [4H,H]
  const float* be_ih = (const float*)d_in[10];
  const float* be_hh = (const float*)d_in[11];
  const float* Wd_ih = (const float*)d_in[12];
  const float* Wd_hh = (const float*)d_in[13];
  const float* bd_ih = (const float*)d_in[14];
  const float* bd_hh = (const float*)d_in[15];
  float* out = (float*)d_out;                       // [B,T,VE] (logits then softmax)

  // ---- workspace bump allocator ----
  char* base = (char*)d_ws;
  size_t off = 0;
  auto alloc = [&](size_t bytes) -> char* {
    off = (off + 255) & ~(size_t)255;
    char* p = base + off;
    off += bytes;
    return p;
  };
  u16* pWe_hi = (u16*)alloc((size_t)Gsz * Hsz * 2);
  u16* pWe_lo = (u16*)alloc((size_t)Gsz * Hsz * 2);
  u16* pWd_hi = (u16*)alloc((size_t)Gsz * Hsz * 2);
  u16* pWd_lo = (u16*)alloc((size_t)Gsz * Hsz * 2);
  u16* pWa_hi = (u16*)alloc((size_t)Hsz * Hsz * 2);
  u16* pWa_lo = (u16*)alloc((size_t)Hsz * Hsz * 2);
  u16* pW1_hi = (u16*)alloc((size_t)Hsz * 2 * Hsz * 2);
  u16* pW1_lo = (u16*)alloc((size_t)Hsz * 2 * Hsz * 2);
  u16* pW2_hi = (u16*)alloc((size_t)Vsz * Hsz * 2);
  u16* pW2_lo = (u16*)alloc((size_t)Vsz * Hsz * 2);
  float* xpe = (float*)alloc((size_t)Ssz * Bsz * Gsz * 4);
  float* xpd = (float*)alloc((size_t)Tsz * Bsz * Gsz * 4);
  float* enc_h = (float*)alloc((size_t)Ssz * Bsz * Hsz * 4);
  float* proj = (float*)alloc((size_t)Ssz * Bsz * Hsz * 4);
  float* dec_h = (float*)alloc((size_t)Tsz * Bsz * Hsz * 4);
  float* c_buf = (float*)alloc((size_t)Bsz * Hsz * 4);
  float* attw = (float*)alloc((size_t)Tsz * Bsz * Ssz * 4);
  float* hc = (float*)alloc((size_t)Tsz * Bsz * 2 * Hsz * 4);
  float* hid = (float*)alloc((size_t)Tsz * Bsz * Hsz * 4);
  (void)ws_size; (void)in_sizes; (void)n_in; (void)out_size;

  auto packGrid = [](int N, int K) { return ((N >> 4) * (K >> 5) * 32 + 255) / 256; };

  // 1) pack weights into WMMA-B bf16 hi/lo layout
  k_pack<<<packGrid(Gsz, Hsz), 256, 0, stream>>>(We_hh, pWe_hi, pWe_lo, Gsz, Hsz);
  k_pack<<<packGrid(Gsz, Hsz), 256, 0, stream>>>(Wd_hh, pWd_hi, pWd_lo, Gsz, Hsz);
  k_pack<<<packGrid(Hsz, Hsz), 256, 0, stream>>>(Wa, pWa_hi, pWa_lo, Hsz, Hsz);
  k_pack<<<packGrid(Hsz, 2 * Hsz), 256, 0, stream>>>(W1, pW1_hi, pW1_lo, Hsz, 2 * Hsz);
  k_pack<<<packGrid(Vsz, Hsz), 256, 0, stream>>>(W2, pW2_hi, pW2_lo, Vsz, Hsz);

  // 2) embedding gathers (+ folded biases)
  k_gather<<<Ssz * Bsz, 256, 0, stream>>>(input_batch, We_ih, be_ih, be_hh, xpe, Ssz);
  k_gather<<<Tsz * Bsz, 256, 0, stream>>>(output_batch, Wd_ih, bd_ih, bd_hh, xpd, Tsz);

  // 3) encoder recurrence (h0=c0=0)
  k_lstm<<<1, 512, 65536, stream>>>(xpe, pWe_hi, pWe_lo, nullptr, nullptr, enc_h, c_buf, Ssz);

  // 4) proj_enc = enc_h @ Wa^T + ba
  {
    int waves = ((Ssz * Bsz) >> 4) * (Hsz >> 6);
    k_gemm<<<(waves + 7) / 8, 256, 0, stream>>>(enc_h, pWa_hi, pWa_lo, ba, proj,
                                                Ssz * Bsz, Hsz, Hsz, 0, 0);
  }

  // 5) decoder recurrence seeded with encoder final (h,c)
  k_lstm<<<1, 512, 65536, stream>>>(xpd, pWd_hi, pWd_lo,
                                    enc_h + (size_t)(Ssz - 1) * Bsz * Hsz, c_buf,
                                    dec_h, c_buf, Tsz);

  // 6) attention (block per batch, enc state resident in 256KB LDS)
  k_attn_scores<<<Bsz, 256, (65536 + 512 + 256 + 128) * 4, stream>>>(proj, dec_h, attw);
  k_attn_ctx<<<Bsz, 256, (65536 + 128) * 4, stream>>>(enc_h, dec_h, attw, hc);

  // 7) hid = tanh(hc @ W1^T + b1)
  {
    int waves = ((Tsz * Bsz) >> 4) * (Hsz >> 6);
    k_gemm<<<(waves + 7) / 8, 256, 0, stream>>>(hc, pW1_hi, pW1_lo, b1, hid,
                                                Tsz * Bsz, Hsz, 2 * Hsz, 1, 0);
  }

  // 8) logits = hid @ W2^T + b2, written transposed straight into d_out
  {
    int waves = ((Tsz * Bsz) >> 4) * (Vsz >> 6);
    k_gemm<<<(waves + 7) / 8, 256, 0, stream>>>(hid, pW2_hi, pW2_lo, b2, out,
                                                Tsz * Bsz, Vsz, Hsz, 0, 1);
  }

  // 9) softmax in place on d_out rows
  k_softmax<<<Bsz * Tsz, 256, 0, stream>>>(out);
}
